// A2aSparseMLP_21861383537337
// MI455X (gfx1250) — matrix operand order repaired and checked by reading, not compiled, not run on hardware
//
#include <hip/hip_runtime.h>
#include <math.h>

#define ALPHA_C 1.702f
#define LIMIT_C 7.0f
#define S_TOK 2048
#define H_DIM 1024
#define E_NUM 32
#define I_DIM 1024
#define TOPK 4
#define NSLOT (S_TOK * TOPK)            // 8192 (token,k) pairs
#define SLOT_MAX (NSLOT + E_NUM * 31)   // 9184 slots after 32-padding per expert
#define TILE_MAX (SLOT_MAX / 32)        // 287 token tiles (32 tokens each)
#define DUMMY_SK NSLOT                  // padding rows write here

typedef __attribute__((ext_vector_type(16))) __bf16 v16bf;
typedef __attribute__((ext_vector_type(8)))  float  v8f;

// ---------------- init: counters, pad defaults, dummy weight ----------------
__global__ void init_kernel(int* counts, int* counts2, int* tile_expert,
                            int* slot_token, int* slot_sk, float* topw) {
    int id = blockIdx.x * 256 + threadIdx.x;
    if (id < E_NUM) { counts[id] = 0; counts2[id] = 0; }
    if (id < TILE_MAX) tile_expert[id] = -1;
    if (id < SLOT_MAX) { slot_token[id] = 0; slot_sk[id] = DUMMY_SK; }
    if (id == 0) topw[DUMMY_SK] = 0.0f;   // zero routing weight for padding rows
}

// ---------------- router: 1 wave per token, lane = expert -------------------
__global__ __launch_bounds__(32)
void router_kernel(const float* __restrict__ hs, const float* __restrict__ rw,
                   const float* __restrict__ rb, float* __restrict__ scores,
                   float* __restrict__ topw, int* __restrict__ topi) {
    int s = blockIdx.x;
    int e = threadIdx.x;                  // 0..31 (wave32)
    const float* x = hs + (size_t)s * H_DIM;
    float acc = 0.f;
#pragma unroll 4
    for (int h = 0; h < H_DIM; ++h)
        acc += x[h] * rw[(size_t)h * E_NUM + e];
    acc += rb[e];

    float v = acc;
    float tv[TOPK]; int ti[TOPK];
#pragma unroll
    for (int k = 0; k < TOPK; ++k) {
        float bv = v; int bi = e;
        for (int off = 16; off > 0; off >>= 1) {
            float ov = __shfl_xor(bv, off, 32);
            int   oi = __shfl_xor(bi, off, 32);
            if (ov > bv || (ov == bv && oi < bi)) { bv = ov; bi = oi; }
        }
        tv[k] = bv; ti[k] = bi;
        if (e == bi) v = -3.4e38f;
    }
    // softmax over the 4 selected (tv[0] is the max)
    float ex[TOPK], sum = 0.f;
#pragma unroll
    for (int k = 0; k < TOPK; ++k) { ex[k] = __expf(tv[k] - tv[0]); sum += ex[k]; }
    float inv = __builtin_amdgcn_rcpf(sum);
    float sc = 0.f;
#pragma unroll
    for (int k = 0; k < TOPK; ++k) if (ti[k] == e) sc = ex[k] * inv;
    scores[(size_t)s * E_NUM + e] = sc;
    if (e < TOPK) {
        topw[s * TOPK + e] = ex[e] * inv;
        topi[s * TOPK + e] = ti[e];
    }
}

// ---------------- expert histogram / scan / compaction ----------------------
__global__ void count_kernel(const int* __restrict__ topi, int* counts) {
    int id = blockIdx.x * 256 + threadIdx.x;
    if (id < NSLOT) atomicAdd(&counts[topi[id]], 1);
}

__global__ __launch_bounds__(32)
void scan_kernel(const int* __restrict__ counts, int* offsets, int* tile_expert) {
    int e = threadIdx.x;                        // 0..31
    int c = counts[e];
    int p = (c + 31) & ~31;                     // pad to 32-token tile multiple
    int x = p;
    for (int off = 1; off < 32; off <<= 1) {
        int y = __shfl_up(x, off, 32);
        if (e >= off) x += y;
    }
    int excl = x - p;
    offsets[e] = excl;
    if (e == 31) offsets[E_NUM] = x;
    int t0 = excl >> 5, nt = p >> 5;
    for (int i = 0; i < nt; ++i) tile_expert[t0 + i] = e;
}

__global__ void fill_kernel(const int* __restrict__ topi, const int* __restrict__ offsets,
                            int* counts2, int* slot_token, int* slot_sk) {
    int id = blockIdx.x * 256 + threadIdx.x;    // id = s*4 + k
    if (id >= NSLOT) return;
    int e = topi[id];
    int pos = atomicAdd(&counts2[e], 1);
    int slot = offsets[e] + pos;
    slot_token[slot] = id >> 2;
    slot_sk[slot] = id;
}

// ---- activation: (u+1) * g * sigmoid(alpha*g), with reference clamps -------
__device__ __forceinline__ float act_fn(float g, float u) {
    g = fminf(g, LIMIT_C);
    u = fminf(fmaxf(u, -LIMIT_C), LIMIT_C);
    float sig = __builtin_amdgcn_rcpf(1.f + __expf(-ALPHA_C * g));
    return (u + 1.f) * g * sig;
}

// ---------------- GEMM 1: gate_up + activation (bf16 WMMA) ------------------
// One wave computes a 32(token) x 16(i) tile: two A fragments share the B pair,
// so each set of 16 B float2-loads feeds 4 WMMAs.
__global__ __launch_bounds__(32)
void gateup_kernel(const float* __restrict__ hs, const float* __restrict__ gup,
                   const float* __restrict__ gub, const int* __restrict__ tile_expert,
                   const int* __restrict__ slot_token, __bf16* __restrict__ act) {
    int bt = blockIdx.y;
    int e = tile_expert[bt];
    if (e < 0) return;
    int i0 = blockIdx.x * 16;
    int L = threadIdx.x, m = L & 15, hl = L >> 4;

    int base = bt * 32;
    int token0 = slot_token[base + m];
    int token1 = slot_token[base + 16 + m];
    const float* xrow0 = hs + (size_t)token0 * H_DIM;
    const float* xrow1 = hs + (size_t)token1 * H_DIM;
    const float* wb = gup + (size_t)e * H_DIM * 2 * I_DIM + 2 * (i0 + m);

    v8f accg0 = {}, accu0 = {}, accg1 = {}, accu1 = {};
    for (int k0 = 0; k0 < H_DIM; k0 += 32) {
        v16bf a0, a1;
        {
            const float* p0a = xrow0 + k0 + 8 * hl;        // K = 8*half + j
            const float* p0b = xrow0 + k0 + 16 + 8 * hl;   // K = 16 + 8*half + j
            const float* p1a = xrow1 + k0 + 8 * hl;
            const float* p1b = xrow1 + k0 + 16 + 8 * hl;
#pragma unroll
            for (int j = 0; j < 8; ++j) {
                a0[j] = (__bf16)p0a[j]; a0[j + 8] = (__bf16)p0b[j];
                a1[j] = (__bf16)p1a[j]; a1[j + 8] = (__bf16)p1b[j];
            }
        }
        v16bf bg, bu;
#pragma unroll
        for (int j = 0; j < 16; ++j) {                     // K = 8*half + j (+8 for j>=8)
            const float* wp = wb + (size_t)(k0 + 8 * hl + j + (j >= 8 ? 8 : 0)) * (2 * I_DIM);
            bg[j] = (__bf16)wp[0];
            bu[j] = (__bf16)wp[1];
        }
        accg0 = __builtin_amdgcn_wmma_f32_16x16x32_bf16(false, a0, false, bg,
                                                        (short)0, accg0, false, false);
        accu0 = __builtin_amdgcn_wmma_f32_16x16x32_bf16(false, a0, false, bu,
                                                        (short)0, accu0, false, false);
        accg1 = __builtin_amdgcn_wmma_f32_16x16x32_bf16(false, a1, false, bg,
                                                        (short)0, accg1, false, false);
        accu1 = __builtin_amdgcn_wmma_f32_16x16x32_bf16(false, a1, false, bu,
                                                        (short)0, accu1, false, false);
    }
    float bgb = gub[(size_t)e * 2 * I_DIM + 2 * (i0 + m)];
    float bub = gub[(size_t)e * 2 * I_DIM + 2 * (i0 + m) + 1];
#pragma unroll
    for (int r = 0; r < 8; ++r) {
        int row = r + 8 * hl;                              // C: M = r + 8*half
        float a0v = act_fn(accg0[r] + bgb, accu0[r] + bub);
        float a1v = act_fn(accg1[r] + bgb, accu1[r] + bub);
        act[(size_t)(base + row) * I_DIM + i0 + m] = (__bf16)a0v;
        act[(size_t)(base + 16 + row) * I_DIM + i0 + m] = (__bf16)a1v;
    }
}

// ---------------- GEMM 2: down proj, weighted write -------------------------
__global__ __launch_bounds__(32)
void down_kernel(const __bf16* __restrict__ act, const float* __restrict__ dp,
                 const float* __restrict__ db, const int* __restrict__ tile_expert,
                 const int* __restrict__ slot_sk, const float* __restrict__ topw,
                 float* __restrict__ tmp) {
    int bt = blockIdx.y;
    int e = tile_expert[bt];
    if (e < 0) return;
    int h0 = blockIdx.x * 16;
    int L = threadIdx.x, m = L & 15, hl = L >> 4;

    int base = bt * 32;
    const __bf16* arow0 = act + (size_t)(base + m) * I_DIM;
    const __bf16* arow1 = act + (size_t)(base + 16 + m) * I_DIM;
    const float* wb = dp + (size_t)e * I_DIM * H_DIM + (h0 + m);

    v8f acc0 = {}, acc1 = {};
    for (int k0 = 0; k0 < I_DIM; k0 += 32) {
        v16bf a0, a1;
        {
            const __bf16* p0a = arow0 + k0 + 8 * hl;
            const __bf16* p0b = arow0 + k0 + 16 + 8 * hl;
            const __bf16* p1a = arow1 + k0 + 8 * hl;
            const __bf16* p1b = arow1 + k0 + 16 + 8 * hl;
#pragma unroll
            for (int j = 0; j < 8; ++j) {
                a0[j] = p0a[j]; a0[j + 8] = p0b[j];
                a1[j] = p1a[j]; a1[j + 8] = p1b[j];
            }
        }
        v16bf b;
#pragma unroll
        for (int j = 0; j < 16; ++j)
            b[j] = (__bf16)wb[(size_t)(k0 + 8 * hl + j + (j >= 8 ? 8 : 0)) * H_DIM];
        acc0 = __builtin_amdgcn_wmma_f32_16x16x32_bf16(false, a0, false, b,
                                                       (short)0, acc0, false, false);
        acc1 = __builtin_amdgcn_wmma_f32_16x16x32_bf16(false, a1, false, b,
                                                       (short)0, acc1, false, false);
    }
    float bias = db[(size_t)e * H_DIM + h0 + m];
#pragma unroll
    for (int r = 0; r < 8; ++r) {
        int row = r + 8 * hl;
        int sk0 = slot_sk[base + row];           // = s*4+k, or DUMMY_SK for padding
        int sk1 = slot_sk[base + 16 + row];
        float w0 = topw[sk0];                    // topw[DUMMY_SK] == 0
        float w1 = topw[sk1];
        tmp[(size_t)sk0 * H_DIM + h0 + m] = w0 * (acc0[r] + bias);
        tmp[(size_t)sk1 * H_DIM + h0 + m] = w1 * (acc1[r] + bias);
    }
}

// ---------------- combine: out[s,h] = sum_k tmp[s*4+k, h] -------------------
__global__ void combine_kernel(const float* __restrict__ tmp, float* __restrict__ out) {
    int id = blockIdx.x * 256 + threadIdx.x;   // over S*H
    int s = id >> 10, h = id & (H_DIM - 1);
    const float* base = tmp + (size_t)(s * TOPK) * H_DIM + h;
    out[id] = base[0] + base[H_DIM] + base[2 * H_DIM] + base[3 * H_DIM];
}

// ---------------- launch ----------------------------------------------------
extern "C" void kernel_launch(void* const* d_in, const int* in_sizes, int n_in,
                              void* d_out, int out_size, void* d_ws, size_t ws_size,
                              hipStream_t stream) {
    const float* hs  = (const float*)d_in[0];
    const float* rw  = (const float*)d_in[1];
    const float* rb  = (const float*)d_in[2];
    const float* gup = (const float*)d_in[3];
    const float* gub = (const float*)d_in[4];
    const float* dp  = (const float*)d_in[5];
    const float* db  = (const float*)d_in[6];

    float* out    = (float*)d_out;                       // [S,H]
    float* scores = out + (size_t)S_TOK * H_DIM;         // [S,E]

    char* w = (char*)d_ws;
    float*  topw        = (float*)(w);                   // (NSLOT+1) floats
    int*    topi        = (int*)  (w + 33024);           // NSLOT ints
    int*    counts      = (int*)  (w + 65792);
    int*    counts2     = (int*)  (w + 65920);
    int*    offsets     = (int*)  (w + 66048);           // E_NUM+1 ints
    int*    tile_expert = (int*)  (w + 66304);           // TILE_MAX ints
    int*    slot_token  = (int*)  (w + 68608);           // SLOT_MAX ints (36736 B)
    int*    slot_sk     = (int*)  (w + 105472);          // SLOT_MAX ints (36736 B)
    __bf16* act         = (__bf16*)(w + 142336);         // SLOT_MAX*I bf16 (~18 MB)
    float*  tmp         = (float*)(w + 142336 + (size_t)SLOT_MAX * I_DIM * 2); // (NSLOT+1)*H f32

    init_kernel<<<(SLOT_MAX + 255) / 256, 256, 0, stream>>>(counts, counts2,
        tile_expert, slot_token, slot_sk, topw);
    router_kernel<<<S_TOK, 32, 0, stream>>>(hs, rw, rb, scores, topw, topi);
    count_kernel<<<NSLOT / 256, 256, 0, stream>>>(topi, counts);
    scan_kernel<<<1, 32, 0, stream>>>(counts, offsets, tile_expert);
    fill_kernel<<<NSLOT / 256, 256, 0, stream>>>(topi, offsets, counts2,
        slot_token, slot_sk);
    gateup_kernel<<<dim3(I_DIM / 16, TILE_MAX), 32, 0, stream>>>(hs, gup, gub,
        tile_expert, slot_token, act);
    down_kernel<<<dim3(H_DIM / 16, TILE_MAX), 32, 0, stream>>>(act, dp, db,
        tile_expert, slot_sk, topw, tmp);
    combine_kernel<<<(S_TOK * H_DIM) / 256, 256, 0, stream>>>(tmp, out);
}
